// KazdovBlock_82360292868517
// MI455X (gfx1250) — matrix operand order repaired and checked by reading, not compiled, not run on hardware
//
#include <hip/hip_runtime.h>
#include <hip/hip_bf16.h>

// ---------------------------------------------------------------------------
// MI455X (gfx1250) implementation.
// All matmuls: V_WMMA_F32_16X16X32_BF16 (bf16 in, f32 accumulate).
// GEMM: 8 waves/block, each wave owns a 16x(16*NG) strip; all waves share the
// same NG B-fragments per K-step, staged through LDS with
// GLOBAL_LOAD_ASYNC_TO_LDS_B128 (double-buffered, ASYNCcnt + barrier).
// A fragments are register-double-buffered from global.
// ---------------------------------------------------------------------------

typedef __bf16 bf16;
typedef __attribute__((ext_vector_type(16))) __bf16 v16bf;
typedef __attribute__((ext_vector_type(8)))  float  v8f;

// Raw LDS byte offset of a __shared__ object (addrspace(3) ptrtoint).
__device__ __forceinline__ unsigned lds_off(const void* p) {
  return (unsigned)(unsigned long long)(__attribute__((address_space(3))) const void*)p;
}

// A-matrix (16x32 bf16, MxK) per-lane K mapping (ISA 7.12.2):
//   lane L: row = L%16, hi = L/16; element e: v=e/2,
//   K = (v>=4?16:0) + hi*8 + 2*(v%4) + (e&1)
__device__ __forceinline__ int kmapA(int e, int hi) {
  int v = e >> 1;
  return ((v & 4) ? 16 : 0) + hi * 8 + ((v & 3) << 1) + (e & 1);
}

// ---------------------------------------------------------------------------
// Pack A: row-major f32 (M x K) -> bf16 WMMA-A fragment layout.
// Tile (mt, kt) at ((mt*ktiles + kt)*512); within tile lane*16 + e.
// ---------------------------------------------------------------------------
__global__ __launch_bounds__(256)
void pack_a_f32(const float* __restrict__ src, bf16* __restrict__ dst,
                int M, int K) {
  size_t idx = (size_t)blockIdx.x * blockDim.x + threadIdx.x;
  size_t total = (size_t)M * K;
  if (idx >= total) return;
  int e    = (int)(idx & 15);
  int lane = (int)((idx >> 4) & 31);
  size_t tile = idx >> 9;
  int ktiles = K >> 5;
  int kt = (int)(tile % ktiles);
  int mt = (int)(tile / ktiles);
  int row = mt * 16 + (lane & 15);
  int k   = kt * 32 + kmapA(e, lane >> 4);
  dst[idx] = (bf16)src[(size_t)row * K + k];
}

// ---------------------------------------------------------------------------
// Pack B: logical (K x N) -> bf16 WMMA-B fragment layout.
// B-fragment mapping: col N = L%16, K = (e>=8?16:0) + (L/16)*8 + (e&7).
// trans==0: B[k,n] = src[k*lds + n];  trans==1: B[k,n] = src[n*lds + k].
// Tile (nt, kt) at ((nt*ktiles + kt)*512).
// ---------------------------------------------------------------------------
__global__ __launch_bounds__(256)
void pack_b_f32(const float* __restrict__ src, bf16* __restrict__ dst,
                int K, int N, int lds, int trans) {
  size_t idx = (size_t)blockIdx.x * blockDim.x + threadIdx.x;
  size_t total = (size_t)K * N;
  if (idx >= total) return;
  int e    = (int)(idx & 15);
  int lane = (int)((idx >> 4) & 31);
  size_t tile = idx >> 9;
  int ktiles = K >> 5;
  int kt = (int)(tile % ktiles);
  int nt = (int)(tile / ktiles);
  int k = kt * 32 + ((e & 8) ? 16 : 0) + (lane >> 4) * 8 + (e & 7);
  int n = nt * 16 + (lane & 15);
  float v = trans ? src[(size_t)n * lds + k] : src[(size_t)k * lds + n];
  dst[idx] = (bf16)v;
}

// ---------------------------------------------------------------------------
// GEMM: C(MxN) f32 = Apk(MxK) @ Bpk(KxN), fragment layout.
// Requires (M/16) % 8 == 0 so all 8 waves of a block share the same ng
// (same NG B fragments) -> B staged once per block via async-to-LDS.
// ---------------------------------------------------------------------------
template <int NG>
__global__ __launch_bounds__(256)
void gemm_bf16_wmma(const bf16* __restrict__ Apk, const bf16* __restrict__ Bpk,
                    float* __restrict__ C, int M, int N, int K) {
  __shared__ bf16 ldsB[2][NG * 512];  // double-buffered: NG B fragments x 1KB

  const int lane = threadIdx.x & 31;
  const int w    = threadIdx.x >> 5;
  const int wid  = blockIdx.x * 8 + w;
  const int mtiles  = M >> 4;
  const int ngroups = N / (16 * NG);
  const int ktiles  = K >> 5;
  if (wid >= mtiles * ngroups) return;  // grids are exact; never taken
  const int mt = wid % mtiles;
  const int ng = wid / mtiles;          // identical for all waves in block

  const bf16* aPtr  = Apk + (size_t)mt * ktiles * 512 + lane * 16;
  const bf16* bBase = Bpk + (size_t)(ng * NG) * ktiles * 512 + lane * 16;
  const size_t bStride = (size_t)ktiles * 512;  // elements between n-tiles

  // Wave 0: async-DMA the NG B fragments of K-step kt into ldsB[buf].
  // Per lane 32B => two b128 ops; INST_OFFSET applies to LDS and global.
  auto stageB = [&](int kt, int buf) {
    unsigned l0 = lds_off(&ldsB[buf][0]) + (unsigned)(lane * 32);
    const bf16* g = bBase + (size_t)kt * 512;
#pragma unroll
    for (int j = 0; j < NG; ++j) {
      unsigned lj = l0 + (unsigned)(j * 1024);
      unsigned long long gj = (unsigned long long)(g + (size_t)j * bStride);
      asm volatile("global_load_async_to_lds_b128 %0, %1, off"
                   :: "v"(lj), "v"(gj) : "memory");
      asm volatile("global_load_async_to_lds_b128 %0, %1, off offset:16"
                   :: "v"(lj), "v"(gj) : "memory");
    }
  };

  if (w == 0) {
    stageB(0, 0);
    asm volatile("s_wait_asynccnt 0x0" ::: "memory");
  }
  __syncthreads();

  v8f acc[NG] = {};
  v16bf a = *(const v16bf*)aPtr;
  aPtr += 512;

  // Steady state: unconditional prefetch of step kt+1 while computing kt.
  for (int kt = 0; kt < ktiles - 1; ++kt) {
    const int buf = kt & 1;
    if (w == 0) stageB(kt + 1, buf ^ 1);
    v16bf aN = *(const v16bf*)aPtr;   // next A fragment (registers)
    aPtr += 512;

#pragma unroll
    for (int j = 0; j < NG; ++j) {
      v16bf b = *(const v16bf*)&ldsB[buf][j * 512 + lane * 16];  // ds_load_b128 x2
      acc[j] = __builtin_amdgcn_wmma_f32_16x16x32_bf16(
          false, a, false, b, (short)0, acc[j], false, false);
    }
    a = aN;

    // Readers' ds_loads retired before their WMMAs (s_wait_dscnt), which
    // precede this barrier; wave 0 publishes the freshly staged buffer.
    if (w == 0) asm volatile("s_wait_asynccnt 0x0" ::: "memory");
    __syncthreads();
  }

  // Final K-step (no staging, no barrier needed afterwards).
  {
    const int buf = (ktiles - 1) & 1;
#pragma unroll
    for (int j = 0; j < NG; ++j) {
      v16bf b = *(const v16bf*)&ldsB[buf][j * 512 + lane * 16];
      acc[j] = __builtin_amdgcn_wmma_f32_16x16x32_bf16(
          false, a, false, b, (short)0, acc[j], false, false);
    }
  }

  // C/D layout: VGPR r -> row (r + 8*hi), col = lane%16.
  const int row0 = mt * 16 + (lane >> 4) * 8;
  const int col  = lane & 15;
#pragma unroll
  for (int j = 0; j < NG; ++j) {
    const int c0 = (ng * NG + j) * 16 + col;
#pragma unroll
    for (int r = 0; r < 8; ++r)
      C[(size_t)(row0 + r) * N + c0] = acc[j][r];
  }
}

// ---------------------------------------------------------------------------
// Causal mean Z[b,t,d] = cumsum(x*m)/max(n,1); also emits n[b,t] (d==0 lane).
// ---------------------------------------------------------------------------
__global__ __launch_bounds__(256)
void zmean_kernel(const float* __restrict__ x, const unsigned char* __restrict__ mask,
                  float* __restrict__ Z, float* __restrict__ nbuf,
                  int Bb, int T, int D) {
  int id = blockIdx.x * blockDim.x + threadIdx.x;
  if (id >= Bb * D) return;
  int b = id / D, d = id % D;
  float n = 0.f, xs = 0.f;
  for (int t = 0; t < T; ++t) {
    size_t r = (size_t)(b * T + t);
    float m = mask[r] ? 1.f : 0.f;
    n += m;
    xs += m * x[r * D + d];
    Z[r * D + d] = xs / fmaxf(n, 1.f);
    if (d == 0) nbuf[r] = n;
  }
}

// Masked causal cumsum over T, per (b,c) column. Safe in-place.
__global__ __launch_bounds__(256)
void cumsum_masked(const float* __restrict__ src, float* __restrict__ dst,
                   const unsigned char* __restrict__ mask, int Bb, int T, int C) {
  int id = blockIdx.x * blockDim.x + threadIdx.x;
  if (id >= Bb * C) return;
  int b = id / C, c = id % C;
  float s = 0.f;
  for (int t = 0; t < T; ++t) {
    size_t r = (size_t)(b * T + t);
    float m = mask[r] ? 1.f : 0.f;
    s += m * src[r * C + c];
    dst[r * C + c] = s;
  }
}

// out = QVb*cKWb + alpha_bi*(iQWb*ciKVb); out may alias QVb.
__global__ __launch_bounds__(256)
void combine_bi(const float* __restrict__ QVb, const float* __restrict__ cKWb,
                const float* __restrict__ iQWb, const float* __restrict__ ciKVb,
                const float* __restrict__ a_bi, float* __restrict__ out, size_t n) {
  size_t i = (size_t)blockIdx.x * blockDim.x + threadIdx.x;
  if (i >= n) return;
  out[i] = QVb[i] * cKWb[i] + a_bi[0] * (iQWb[i] * ciKVb[i]);
}

// out = a_tri*QVt*ZXt*cKWt; out may alias QVt.
__global__ __launch_bounds__(256)
void combine_tri(const float* __restrict__ QVt, const float* __restrict__ ZXt,
                 const float* __restrict__ cKWt, const float* __restrict__ a_tri,
                 float* __restrict__ out, size_t n) {
  size_t i = (size_t)blockIdx.x * blockDim.x + threadIdx.x;
  if (i >= n) return;
  out[i] = a_tri[0] * QVt[i] * ZXt[i] * cKWt[i];
}

// Fused final combine + A-pack:
//   S[row,k] = (P1+P2)/nc + (bias_b*(1+a_bi) + a_tri*bias_t) * ratio
// written directly in bf16 WMMA-A fragment layout for the @Wo.T GEMM.
__global__ __launch_bounds__(256)
void pack_final(const float* __restrict__ P1, const float* __restrict__ P2,
                const float* __restrict__ nbuf,
                const float* __restrict__ bias_b, const float* __restrict__ bias_t,
                const float* __restrict__ a_bi, const float* __restrict__ a_tri,
                bf16* __restrict__ dst, int M, int D) {
  size_t idx = (size_t)blockIdx.x * blockDim.x + threadIdx.x;
  size_t total = (size_t)M * D;
  if (idx >= total) return;
  int e    = (int)(idx & 15);
  int lane = (int)((idx >> 4) & 31);
  size_t tile = idx >> 9;
  int ktiles = D >> 5;
  int kt = (int)(tile % ktiles);
  int mt = (int)(tile / ktiles);
  int row = mt * 16 + (lane & 15);
  int k   = kt * 32 + kmapA(e, lane >> 4);
  float n  = nbuf[row];
  float nc = fmaxf(n, 1.f);
  float ratio = n / nc;
  float bias = bias_b[k] * (1.f + a_bi[0]) + a_tri[0] * bias_t[k];
  size_t o = (size_t)row * D + k;
  float v = (P1[o] + P2[o]) / nc + bias * ratio;
  dst[idx] = (bf16)v;
}

// ---------------------------------------------------------------------------
extern "C" void kernel_launch(void* const* d_in, const int* in_sizes, int n_in,
                              void* d_out, int out_size, void* d_ws, size_t ws_size,
                              hipStream_t stream) {
  (void)in_sizes; (void)n_in; (void)out_size;
  constexpr int Bb = 4, T = 2048, D = 1024, R = 256;
  constexpr int M = Bb * T;  // 8192 rows

  const float*         x      = (const float*)d_in[0];
  const unsigned char* mask   = (const unsigned char*)d_in[1];
  const float*         Wq     = (const float*)d_in[2];
  const float*         Wk     = (const float*)d_in[3];
  const float*         Wo     = (const float*)d_in[4];
  const float*         Winv   = (const float*)d_in[5];
  const float*         U_b    = (const float*)d_in[6];
  const float*         V_b    = (const float*)d_in[7];
  const float*         W_b    = (const float*)d_in[8];
  const float*         bias_b = (const float*)d_in[9];
  const float*         U_t    = (const float*)d_in[10];
  const float*         V_t    = (const float*)d_in[11];
  const float*         W_t    = (const float*)d_in[12];
  const float*         X_t    = (const float*)d_in[13];
  const float*         bias_t = (const float*)d_in[14];
  const float*         a_bi   = (const float*)d_in[15];
  const float*         a_tri  = (const float*)d_in[16];
  float* out = (float*)d_out;

  // Bump allocator over workspace.
  char* base = (char*)d_ws;
  size_t off = 0;
  auto alloc = [&](size_t bytes) -> void* {
    void* r = base + off;
    off += (bytes + 255) & ~(size_t)255;
    return r;
  };

  bf16* wqT  = (bf16*)alloc((size_t)D * D * 2);
  bf16* wkT  = (bf16*)alloc((size_t)D * D * 2);
  bf16* wiT  = (bf16*)alloc((size_t)D * D * 2);
  bf16* woT  = (bf16*)alloc((size_t)D * D * 2);
  bf16* ubT  = (bf16*)alloc((size_t)R * D * 2);
  bf16* utT  = (bf16*)alloc((size_t)R * D * 2);
  bf16* vbP  = (bf16*)alloc((size_t)D * R * 2);
  bf16* wbP  = (bf16*)alloc((size_t)D * R * 2);
  bf16* vtP  = (bf16*)alloc((size_t)D * R * 2);
  bf16* wtP  = (bf16*)alloc((size_t)D * R * 2);
  bf16* xtP  = (bf16*)alloc((size_t)D * R * 2);
  bf16* xpk  = (bf16*)alloc((size_t)M * D * 2);
  bf16* qpk  = (bf16*)alloc((size_t)M * D * 2);
  bf16* kpk  = (bf16*)alloc((size_t)M * D * 2);
  bf16* iqpk = (bf16*)alloc((size_t)M * D * 2);
  bf16* ikpk = (bf16*)alloc((size_t)M * D * 2);
  bf16* zpk  = (bf16*)alloc((size_t)M * D * 2);
  bf16* s1pk = (bf16*)alloc((size_t)M * R * 2);
  bf16* gtpk = (bf16*)alloc((size_t)M * R * 2);
  bf16* spk  = (bf16*)alloc((size_t)M * D * 2);
  float* tmpA = (float*)alloc((size_t)M * D * 4);  // Q,K,invQ,invK,Z,P1 staging
  float* tmpB = (float*)alloc((size_t)M * D * 4);  // P2
  float* QVb  = (float*)alloc((size_t)M * R * 4);
  float* KWb  = (float*)alloc((size_t)M * R * 4);
  float* iQWb = (float*)alloc((size_t)M * R * 4);
  float* iKVb = (float*)alloc((size_t)M * R * 4);
  float* QVt  = (float*)alloc((size_t)M * R * 4);
  float* ZXt  = (float*)alloc((size_t)M * R * 4);
  float* KWt  = (float*)alloc((size_t)M * R * 4);
  float* nbuf = (float*)alloc((size_t)M * 4);
  if (off > ws_size) return;  // workspace too small: safe no-op

  auto blks = [](size_t n) { return (unsigned)((n + 255) / 256); };
  // 16x128 strips (NG=8) when N % 128 == 0 (N=1024 and N=256 alike), else 16x64 (NG=4).
  auto gemm = [&](const bf16* A, const bf16* Bm, float* C, int m, int n, int k) {
    if (n % 128 == 0) {
      int waves = (m >> 4) * (n >> 7);
      gemm_bf16_wmma<8><<<(waves + 7) / 8, 256, 0, stream>>>(A, Bm, C, m, n, k);
    } else {
      int waves = (m >> 4) * (n >> 6);
      gemm_bf16_wmma<4><<<(waves + 7) / 8, 256, 0, stream>>>(A, Bm, C, m, n, k);
    }
  };

  // --- 1. Pack weights (bf16, fragment layout) ---
  pack_b_f32<<<blks((size_t)D * D), 256, 0, stream>>>(Wq,   wqT, D, D, D, 1);
  pack_b_f32<<<blks((size_t)D * D), 256, 0, stream>>>(Wk,   wkT, D, D, D, 1);
  pack_b_f32<<<blks((size_t)D * D), 256, 0, stream>>>(Winv, wiT, D, D, D, 1);
  pack_b_f32<<<blks((size_t)D * D), 256, 0, stream>>>(Wo,   woT, D, D, D, 1);
  pack_b_f32<<<blks((size_t)R * D), 256, 0, stream>>>(U_b,  ubT, R, D, R, 1);
  pack_b_f32<<<blks((size_t)R * D), 256, 0, stream>>>(U_t,  utT, R, D, R, 1);
  pack_b_f32<<<blks((size_t)D * R), 256, 0, stream>>>(V_b,  vbP, D, R, R, 0);
  pack_b_f32<<<blks((size_t)D * R), 256, 0, stream>>>(W_b,  wbP, D, R, R, 0);
  pack_b_f32<<<blks((size_t)D * R), 256, 0, stream>>>(V_t,  vtP, D, R, R, 0);
  pack_b_f32<<<blks((size_t)D * R), 256, 0, stream>>>(W_t,  wtP, D, R, R, 0);
  pack_b_f32<<<blks((size_t)D * R), 256, 0, stream>>>(X_t,  xtP, D, R, R, 0);

  // --- 2. Projections Q, K, invQ, invK (WMMA) ---
  pack_a_f32<<<blks((size_t)M * D), 256, 0, stream>>>(x, xpk, M, D);
  gemm(xpk, wqT, tmpA, M, D, D);                                       // Q
  pack_a_f32<<<blks((size_t)M * D), 256, 0, stream>>>(tmpA, qpk, M, D);
  gemm(xpk, wkT, tmpA, M, D, D);                                       // K
  pack_a_f32<<<blks((size_t)M * D), 256, 0, stream>>>(tmpA, kpk, M, D);
  gemm(qpk, wiT, tmpA, M, D, D);                                       // invQ
  pack_a_f32<<<blks((size_t)M * D), 256, 0, stream>>>(tmpA, iqpk, M, D);
  gemm(kpk, wiT, tmpA, M, D, D);                                       // invK
  pack_a_f32<<<blks((size_t)M * D), 256, 0, stream>>>(tmpA, ikpk, M, D);

  // --- 3. Causal mean Z and valid-count n ---
  zmean_kernel<<<blks((size_t)Bb * D), 256, 0, stream>>>(x, mask, tmpA, nbuf, Bb, T, D);
  pack_a_f32<<<blks((size_t)M * D), 256, 0, stream>>>(tmpA, zpk, M, D);

  // --- 4. Rank-space projections (WMMA) ---
  gemm(qpk,  vbP, QVb,  M, R, D);
  gemm(kpk,  wbP, KWb,  M, R, D);
  gemm(iqpk, wbP, iQWb, M, R, D);
  gemm(ikpk, vbP, iKVb, M, R, D);
  gemm(qpk,  vtP, QVt,  M, R, D);
  gemm(zpk,  xtP, ZXt,  M, R, D);
  gemm(kpk,  wtP, KWt,  M, R, D);

  // --- 5. Masked causal cumsums (in-place) ---
  cumsum_masked<<<blks((size_t)Bb * R), 256, 0, stream>>>(KWb,  KWb,  mask, Bb, T, R);
  cumsum_masked<<<blks((size_t)Bb * R), 256, 0, stream>>>(iKVb, iKVb, mask, Bb, T, R);
  cumsum_masked<<<blks((size_t)Bb * R), 256, 0, stream>>>(KWt,  KWt,  mask, Bb, T, R);

  // --- 6. Rank-space combines (alphas folded in), pack for rank->D GEMMs ---
  combine_bi<<<blks((size_t)M * R), 256, 0, stream>>>(QVb, KWb, iQWb, iKVb, a_bi, QVb, (size_t)M * R);
  combine_tri<<<blks((size_t)M * R), 256, 0, stream>>>(QVt, ZXt, KWt, a_tri, QVt, (size_t)M * R);
  pack_a_f32<<<blks((size_t)M * R), 256, 0, stream>>>(QVb, s1pk, M, R);
  pack_a_f32<<<blks((size_t)M * R), 256, 0, stream>>>(QVt, gtpk, M, R);

  // --- 7. Rank -> D (WMMA), final combine, output projection (WMMA) ---
  gemm(s1pk, ubT, tmpA, M, D, R);   // (fwd + a_bi*inv) @ U_b.T
  gemm(gtpk, utT, tmpB, M, D, R);   // a_tri * tri @ U_t.T
  pack_final<<<blks((size_t)M * D), 256, 0, stream>>>(tmpA, tmpB, nbuf, bias_b, bias_t,
                                                      a_bi, a_tri, spk, M, D);
  gemm(spk, woT, out, M, D, D);     // @ Wo.T
}
